// LightSelfAttention_25031069401608
// MI455X (gfx1250) — matrix operand order repaired and checked by reading, not compiled, not run on hardware
//
#include <hip/hip_runtime.h>
#include <hip/hip_bf16.h>

// ---------------------------------------------------------------------------
// LightSelfAttention for MI455X (gfx1250).
//   k0a: x -> clamp(+-1e4) -> f16          (one pass, removes per-tile cvts)
//   k0b: Wq/Wk/Wv -> f16, Wo -> bf16       (one pass; weights stay hot in L2)
//   k1 : Q/K/V = Xh @ Wh^T + b (f16 WMMA). Q pre-scaled by 0.125*log2(e) so
//        attention softmax runs in base-2. V stored transposed [B,H,D,N].
//   k2 : flash attention, St = K Q^T (transposed scores -> softmax rows are
//        per-lane; Pt C-fragment feeds B operand of Ot = Vt*Pt with only a
//        lane^16 exchange). exp2 via v_exp_f32. Output rounded to bf16.
//   k3 : out = bf16(attn) @ Wo^T + bo (bf16 WMMA, fp32 out).
// ---------------------------------------------------------------------------

typedef __attribute__((ext_vector_type(16))) _Float16 v16h;
typedef __attribute__((ext_vector_type(8)))  _Float16 v8h;
typedef __attribute__((ext_vector_type(16))) __bf16   v16bf;
typedef __attribute__((ext_vector_type(8)))  __bf16   v8bf;
typedef __attribute__((ext_vector_type(8)))  float    v8f;
typedef __attribute__((ext_vector_type(4)))  float    v4f;

#define DEV_INLINE __device__ __forceinline__

static constexpr int   kN = 4096;
static constexpr int   kC = 256;
static constexpr int   kH = 4;
static constexpr int   kD = 64;
// 0.125 * log2(e): folded into Q so softmax is exp2-based.
static constexpr float kQScale = 0.18033688011112042f;

DEV_INLINE v16h cat16h(v8h a, v8h b) {
  return __builtin_shufflevector(a, b, 0,1,2,3,4,5,6,7,8,9,10,11,12,13,14,15);
}
DEV_INLINE v16bf cat16bf(v8bf a, v8bf b) {
  return __builtin_shufflevector(a, b, 0,1,2,3,4,5,6,7,8,9,10,11,12,13,14,15);
}
DEV_INLINE v8f wmma_h(v16h a, v16h b, v8f c) {
  return __builtin_amdgcn_wmma_f32_16x16x32_f16(false, a, false, b, (short)0, c, false, false);
}
DEV_INLINE v8f wmma_bf(v16bf a, v16bf b, v8f c) {
  return __builtin_amdgcn_wmma_f32_16x16x32_bf16(false, a, false, b, (short)0, c, false, false);
}

// ---------------------------------------------------------------------------
// k0a: clamp + convert x (fp32 [2*4096, 256]) to f16. 8 elems / thread.
// ---------------------------------------------------------------------------
__global__ __launch_bounds__(256) void cvt_x_kernel(
    const float* __restrict__ x, _Float16* __restrict__ Xh) {
  const size_t i = ((size_t)blockIdx.x * blockDim.x + threadIdx.x) * 8;
  v4f a = *(const v4f*)(x + i);
  v4f b = *(const v4f*)(x + i + 4);
  v8h r;
#pragma unroll
  for (int j = 0; j < 4; ++j) {
    r[j]     = (_Float16)fminf(fmaxf(a[j], -10000.f), 10000.f);
    r[j + 4] = (_Float16)fminf(fmaxf(b[j], -10000.f), 10000.f);
  }
  *(v8h*)(Xh + i) = r;
}

// ---------------------------------------------------------------------------
// k0b: convert 256x256 weights: Wq/Wk/Wv -> f16, Wo -> bf16.
// 32 blocks x 256 threads x 8 elems = 65536 elems per weight.
// ---------------------------------------------------------------------------
__global__ __launch_bounds__(256) void cvt_w_kernel(
    const float* __restrict__ Wq, const float* __restrict__ Wk,
    const float* __restrict__ Wv, const float* __restrict__ Wo,
    _Float16* __restrict__ Wqh, _Float16* __restrict__ Wkh,
    _Float16* __restrict__ Wvh, __bf16* __restrict__ Wob) {
  const int which = blockIdx.x >> 5;
  const size_t i = (((size_t)(blockIdx.x & 31)) * blockDim.x + threadIdx.x) * 8;
  if (which < 3) {
    const float* src = (which == 0) ? Wq : (which == 1) ? Wk : Wv;
    _Float16*    dst = (which == 0) ? Wqh : (which == 1) ? Wkh : Wvh;
    v4f a = *(const v4f*)(src + i);
    v4f b = *(const v4f*)(src + i + 4);
    v8h r;
#pragma unroll
    for (int j = 0; j < 4; ++j) { r[j] = (_Float16)a[j]; r[j + 4] = (_Float16)b[j]; }
    *(v8h*)(dst + i) = r;
  } else {
    v4f a = *(const v4f*)(Wo + i);
    v4f b = *(const v4f*)(Wo + i + 4);
    v8bf r;
#pragma unroll
    for (int j = 0; j < 4; ++j) { r[j] = (__bf16)a[j]; r[j + 4] = (__bf16)b[j]; }
    *(v8bf*)(Wob + i) = r;
  }
}

// ---------------------------------------------------------------------------
// k1: QKV projection, one wave per 16-row tile (512 tiles). Pure f16 loads.
// mat loop fully unrolled so every load keeps a provably-global base
// (no pointer alloca -> global_load, not flat_load).
// ---------------------------------------------------------------------------
__global__ __launch_bounds__(256) void qkv_kernel(
    const _Float16* __restrict__ Xh,
    const _Float16* __restrict__ Wqh, const _Float16* __restrict__ Wkh,
    const _Float16* __restrict__ Wvh,
    const float* __restrict__ bq, const float* __restrict__ bk,
    const float* __restrict__ bv,
    _Float16* __restrict__ Qh, _Float16* __restrict__ Kh,
    _Float16* __restrict__ Vt) {
  const int lane = threadIdx.x & 31;
  const int wave = threadIdx.x >> 5;
  const int mt   = blockIdx.x * (blockDim.x >> 5) + wave;  // 0..511
  const int col  = lane & 15;
  const int hi   = (lane >> 4) & 1;

  // Preload all 8 k-step A fragments (reused for 48 output tiles).
  v16h a[8];
#pragma unroll
  for (int kc = 0; kc < 8; ++kc) {
    const _Float16* p = Xh + (size_t)(mt * 16 + col) * kC + kc * 32 + (hi ? 8 : 0);
    a[kc] = cat16h(*(const v8h*)p, *(const v8h*)(p + 16));
  }

  const int bIdx  = (mt * 16) / kN;       // tile never straddles a batch
  const int nrow0 = (mt * 16) & (kN - 1);

#pragma unroll
  for (int mat = 0; mat < 3; ++mat) {
    const _Float16* W = (mat == 0) ? Wqh : (mat == 1) ? Wkh : Wvh;
    const float*    bb = (mat == 0) ? bq : (mat == 1) ? bk : bv;
    for (int ct = 0; ct < 16; ++ct) {
      const int n = ct * 16 + col;                     // output column per lane
      const _Float16* wrow = W + (size_t)n * kC + hi * 8;
      v8f c = {};
#pragma unroll
      for (int kc = 0; kc < 8; ++kc) {
        const _Float16* p = wrow + kc * 32;
        v16h bf = cat16h(*(const v8h*)p, *(const v8h*)(p + 16));
        c = wmma_h(a[kc], bf, c);
      }
      const float bias = bb[n];
      const int h = n >> 6, d = n & 63;
      if (mat == 0) {
        // Q, pre-scaled for base-2 softmax. [B,H,N,D]
        _Float16* dst = Qh + ((size_t)(bIdx * kH + h) * kN + nrow0 + 8 * hi) * kD + d;
#pragma unroll
        for (int i = 0; i < 8; ++i)
          dst[(size_t)i * kD] = (_Float16)((c[i] + bias) * kQScale);
      } else if (mat == 1) {
        _Float16* dst = Kh + ((size_t)(bIdx * kH + h) * kN + nrow0 + 8 * hi) * kD + d;
#pragma unroll
        for (int i = 0; i < 8; ++i) dst[(size_t)i * kD] = (_Float16)(c[i] + bias);
      } else {
        // Vt [B,H,D,N]: contiguous in N per lane -> one 16B store
        v8h hv;
#pragma unroll
        for (int i = 0; i < 8; ++i) hv[i] = (_Float16)(c[i] + bias);
        _Float16* dst = Vt + ((size_t)(bIdx * kH + h) * kD + d) * kN + nrow0 + 8 * hi;
        *(v8h*)dst = hv;
      }
    }
  }
}

// ---------------------------------------------------------------------------
// k2: flash attention per (b, h, 16-row q-tile); one wave per tile.
// ---------------------------------------------------------------------------
__global__ __launch_bounds__(256) void attn_kernel(
    const _Float16* __restrict__ Qh, const _Float16* __restrict__ Kh,
    const _Float16* __restrict__ Vt, __bf16* __restrict__ Sb) {
  const int lane = threadIdx.x & 31;
  const int wave = threadIdx.x >> 5;
  const int unit = blockIdx.x * (blockDim.x >> 5) + wave;  // 0..2047
  const int b  = unit >> 10;
  const int h  = (unit >> 8) & 3;
  const int mt = unit & 255;
  const int col = lane & 15;   // m within q-tile (C/D column), also A row
  const int hi  = (lane >> 4) & 1;

  const _Float16* Qp = Qh + (size_t)(b * kH + h) * kN * kD;
  const _Float16* Kp = Kh + (size_t)(b * kH + h) * kN * kD;
  const _Float16* Vp = Vt + (size_t)(b * kH + h) * kD * kN;

  // B operand of St: lane reads Q row m=col along d (two 8-f16 runs at d = 8*hi and 16 + 8*hi).
  v16h qb[2];
#pragma unroll
  for (int kc = 0; kc < 2; ++kc) {
    const _Float16* p = Qp + (size_t)(mt * 16 + col) * kD + kc * 32 + hi * 8;
    qb[kc] = cat16h(*(const v8h*)p, *(const v8h*)(p + 16));
  }

  v8f acc[4];
#pragma unroll
  for (int t = 0; t < 4; ++t) acc[t] = (v8f)0.f;
  float rmax = -1e30f, rsum = 0.f;

  for (int jb = 0; jb < kN; jb += 32) {
    // Prefetch K and V lines two chunks ahead (gfx1250 global_prefetch_b8).
    if (jb + 64 < kN) {
      __builtin_prefetch(Kp + (size_t)(jb + 64 + col) * kD, 0, 1);
      __builtin_prefetch(Vp + (size_t)col * kN + jb + 64, 0, 1);
    }
    // --- two transposed score tiles: St[j, m] (already in log2 domain) ---
    v8f st[2];
#pragma unroll
    for (int t = 0; t < 2; ++t) {
      v8f c = {};
#pragma unroll
      for (int kc = 0; kc < 2; ++kc) {
        const _Float16* p =
            Kp + (size_t)(jb + t * 16 + col) * kD + kc * 32 + (hi ? 8 : 0);
        v16h af = cat16h(*(const v8h*)p, *(const v8h*)(p + 16));
        c = wmma_h(af, qb[kc], c);
      }
      st[t] = c;
    }
    // --- online softmax in base 2; j striped over VGPRs + lane^16 ---
    float tmax = st[0][0];
#pragma unroll
    for (int t = 0; t < 2; ++t)
#pragma unroll
      for (int i = 0; i < 8; ++i) tmax = fmaxf(tmax, st[t][i]);
    tmax = fmaxf(tmax, __shfl_xor(tmax, 16, 32));

    if (__any(tmax > rmax)) {
      const float nmax = fmaxf(rmax, tmax);
      const float corr = __builtin_amdgcn_exp2f(rmax - nmax);
      rmax = nmax;
      rsum *= corr;
#pragma unroll
      for (int t = 0; t < 4; ++t)
#pragma unroll
        for (int i = 0; i < 8; ++i) acc[t][i] *= corr;
    }

    float psum = 0.f;
#pragma unroll
    for (int t = 0; t < 2; ++t)
#pragma unroll
      for (int i = 0; i < 8; ++i) {
        st[t][i] = __builtin_amdgcn_exp2f(st[t][i] - rmax);
        psum += st[t][i];
      }
    psum += __shfl_xor(psum, 16, 32);
    rsum += psum;

    // --- build Pt B-fragment: lane^16 exchange merges the j halves ---
    v16h pb;
#pragma unroll
    for (int i = 0; i < 8; ++i) {
      const float x0 = __shfl_xor(st[0][i], 16, 32);
      const float x1 = __shfl_xor(st[1][i], 16, 32);
      pb[i]     = (_Float16)(hi ? x1 : st[0][i]);
      pb[i + 8] = (_Float16)(hi ? st[1][i] : x0);
    }
    // --- Ot[d, m] += Vt[d, j] * Pt[j, m] over this 32-j chunk ---
#pragma unroll
    for (int t = 0; t < 4; ++t) {
      const _Float16* p = Vp + (size_t)(t * 16 + col) * kN + jb + hi * 16;
      v16h af = cat16h(*(const v8h*)p, *(const v8h*)(p + 8));
      acc[t] = wmma_h(af, pb, acc[t]);
    }
  }

  // Finalize: divide by row sum, round to bf16 (reference casts here),
  // store into attn-out [B,N,C]: 8 contiguous bf16 per lane per d-tile.
  const float inv = 1.f / rsum;
#pragma unroll
  for (int t = 0; t < 4; ++t) {
    v8bf ov;
#pragma unroll
    for (int i = 0; i < 8; ++i) ov[i] = (__bf16)(acc[t][i] * inv);
    __bf16* dst = Sb + ((size_t)(b * kN + mt * 16 + col) * kC) + h * kD +
                  t * 16 + hi * 8;
    *(v8bf*)dst = ov;
  }
}

// ---------------------------------------------------------------------------
// k3: out = bf16(attn) @ Wo^T + bo (bf16 WMMA, fp32 result). Pure bf16 loads.
// ---------------------------------------------------------------------------
__global__ __launch_bounds__(256) void oproj_kernel(
    const __bf16* __restrict__ Sb, const __bf16* __restrict__ Wob,
    const float* __restrict__ bo, float* __restrict__ out) {
  const int lane = threadIdx.x & 31;
  const int wave = threadIdx.x >> 5;
  const int mt   = blockIdx.x * (blockDim.x >> 5) + wave;  // 0..511
  const int col  = lane & 15;
  const int hi   = (lane >> 4) & 1;

  v16bf a[8];
#pragma unroll
  for (int kc = 0; kc < 8; ++kc) {
    const __bf16* p = Sb + (size_t)(mt * 16 + col) * kC + kc * 32 + (hi ? 8 : 0);
    a[kc] = cat16bf(*(const v8bf*)p, *(const v8bf*)(p + 16));
  }

  for (int ct = 0; ct < 16; ++ct) {
    const int n = ct * 16 + col;
    const __bf16* wrow = Wob + (size_t)n * kC + hi * 8;
    v8f c = {};
#pragma unroll
    for (int kc = 0; kc < 8; ++kc) {
      const __bf16* p = wrow + kc * 32;
      v16bf bf = cat16bf(*(const v8bf*)p, *(const v8bf*)(p + 16));
      c = wmma_bf(a[kc], bf, c);
    }
    const float bias = bo[n];
    float* dst = out + (size_t)(mt * 16 + 8 * hi) * kC + n;
#pragma unroll
    for (int i = 0; i < 8; ++i) dst[(size_t)i * kC] = c[i] + bias;
  }
}

// ---------------------------------------------------------------------------
extern "C" void kernel_launch(void* const* d_in, const int* in_sizes, int n_in,
                              void* d_out, int out_size, void* d_ws,
                              size_t ws_size, hipStream_t stream) {
  (void)in_sizes; (void)n_in; (void)out_size; (void)ws_size;
  const float* x  = (const float*)d_in[0];
  const float* Wq = (const float*)d_in[1];
  const float* bq = (const float*)d_in[2];
  const float* Wk = (const float*)d_in[3];
  const float* bk = (const float*)d_in[4];
  const float* Wv = (const float*)d_in[5];
  const float* bv = (const float*)d_in[6];
  const float* Wo = (const float*)d_in[7];
  const float* bo = (const float*)d_in[8];
  float* out = (float*)d_out;

  char* ws = (char*)d_ws;
  const size_t MB = 1024 * 1024;
  _Float16* Xh  = (_Float16*)(ws);                    // 4 MB
  _Float16* Qh  = (_Float16*)(ws + 4  * MB);          // 4 MB
  _Float16* Kh  = (_Float16*)(ws + 8  * MB);          // 4 MB
  _Float16* Vt  = (_Float16*)(ws + 12 * MB);          // 4 MB
  __bf16*   Sb  = (__bf16*)  (ws + 16 * MB);          // 4 MB
  _Float16* Wqh = (_Float16*)(ws + 20 * MB);          // 128 KB
  _Float16* Wkh = (_Float16*)(ws + 20 * MB + 128 * 1024);
  _Float16* Wvh = (_Float16*)(ws + 20 * MB + 256 * 1024);
  __bf16*   Wob = (__bf16*)  (ws + 20 * MB + 384 * 1024);

  cvt_x_kernel<<<1024, 256, 0, stream>>>(x, Xh);           // 2M elems / 8
  cvt_w_kernel<<<128, 256, 0, stream>>>(Wq, Wk, Wv, Wo, Wqh, Wkh, Wvh, Wob);
  qkv_kernel<<<64, 256, 0, stream>>>(Xh, Wqh, Wkh, Wvh, bq, bk, bv, Qh, Kh, Vt);
  attn_kernel<<<256, 256, 0, stream>>>(Qh, Kh, Vt, Sb);
  oproj_kernel<<<64, 256, 0, stream>>>(Sb, Wob, bo, out);
}
